// _RFFTDirectWithGMAC_19490561590128
// MI455X (gfx1250) — matrix operand order repaired and checked by reading, not compiled, not run on hardware
//
#include <hip/hip_runtime.h>

typedef __attribute__((ext_vector_type(16))) __bf16 v16bf;
typedef __attribute__((ext_vector_type(8)))  __bf16 v8bf;
typedef __attribute__((ext_vector_type(8)))  float  v8f;

#define NFFT   512
#define NRFFT  257
#define NPAD   272          // 17 * 16 columns per matrix (zero padded)
#define MTOT   128000       // 32 * 4000
#define MTILE  64
#define LDSROW 520          // hi section: 512 + 8 halfword pad
#define RSTRIDE 1040        // full row: [hi 520 | lo 520] halfwords = 2080 B

static __device__ __forceinline__ unsigned short f32_to_bf16(float f) {
  unsigned int u = __float_as_uint(f);
  u += 0x7fffu + ((u >> 16) & 1u);          // round-to-nearest-even
  return (unsigned short)(u >> 16);
}
static __device__ __forceinline__ float bf16_to_f32(unsigned short h) {
  return __uint_as_float(((unsigned int)h) << 16);
}

// Split the two DFT matrices into bf16 hi/lo, zero-padding N 257 -> 272.
__global__ __launch_bounds__(256) void split_weights(
    const float* __restrict__ mr, const float* __restrict__ mi,
    unsigned short* __restrict__ whi, unsigned short* __restrict__ wlo) {
  int idx = blockIdx.x * 256 + threadIdx.x;       // 0 .. 2*272*512-1
  int k   = idx & (NFFT - 1);
  int n   = (idx >> 9) % NPAD;
  int mat = idx / (NPAD * NFFT);
  float v = 0.0f;
  if (n < NRFFT) v = (mat == 0 ? mr : mi)[n * NFFT + k];
  unsigned short hi = f32_to_bf16(v);
  unsigned short lo = f32_to_bf16(v - bf16_to_f32(hi));
  whi[idx] = hi;
  wlo[idx] = lo;
}

static __device__ __forceinline__ v16bf load_frag2(const unsigned short* p0,
                                                   const unsigned short* p1) {
  v16bf r;
  ((v8bf*)&r)[0] = *((const v8bf*)p0);   // 16B chunk -> VGPRs 0..3
  ((v8bf*)&r)[1] = *((const v8bf*)p1);   // 16B chunk -> VGPRs 4..7
  return r;
}

__global__ __launch_bounds__(256) void rfft_gemm(
    const float* __restrict__ x,
    const unsigned short* __restrict__ whi,
    const unsigned short* __restrict__ wlo,
    float* __restrict__ out) {
  extern __shared__ unsigned short smem[];   // [64 rows][hi 520 | lo 520]

  const int tid = threadIdx.x;
  const long m0 = (long)blockIdx.x * MTILE;

  // ---- Phase 1: load 64x512 f32 x-tile, split into bf16 hi/lo in LDS ----
  {
    const int row = tid >> 2;   // 0..63
    const int seg = tid & 3;    // 0..3, 128 floats each
    const float4* src = (const float4*)(x + (m0 + row) * NFFT + seg * 128);
    unsigned short* dhi = smem + row * RSTRIDE + seg * 128;
    unsigned short* dlo = dhi + LDSROW;
#pragma unroll 4
    for (int i = 0; i < 32; ++i) {
      float4 v = src[i];
      unsigned short h0 = f32_to_bf16(v.x), h1 = f32_to_bf16(v.y);
      unsigned short h2 = f32_to_bf16(v.z), h3 = f32_to_bf16(v.w);
      unsigned short l0 = f32_to_bf16(v.x - bf16_to_f32(h0));
      unsigned short l1 = f32_to_bf16(v.y - bf16_to_f32(h1));
      unsigned short l2 = f32_to_bf16(v.z - bf16_to_f32(h2));
      unsigned short l3 = f32_to_bf16(v.w - bf16_to_f32(h3));
      *(ushort4*)(dhi + i * 4) = make_ushort4(h0, h1, h2, h3);
      *(ushort4*)(dlo + i * 4) = make_ushort4(l0, l1, l2, l3);
    }
  }
  __syncthreads();

  // ---- Phase 2: per-wave WMMA GEMM ----
  const int wave   = tid >> 5;
  const int lane   = tid & 31;
  const int lane15 = lane & 15;
  const int half   = lane >> 4;

  // Loop-invariant per-ms LDS row bases: all inner ds_loads become base+imm.
  const unsigned short* arow[4];
#pragma unroll
  for (int ms = 0; ms < 4; ++ms)
    arow[ms] = smem + (ms * 16 + lane15) * RSTRIDE;

  const int bkoff = 16 * half;       // B: lane half selects K block
  const int akoff = 16 * half;       // A: lane half selects K block

  for (int t = wave; t < 34; t += 8) {       // 34 N-tiles: 17 per matrix
    const int mat = t / 17;
    const int nc0 = (t - mat * 17) * 16;
    const unsigned short* bh_row =
        whi + (size_t)(mat * NPAD + nc0 + lane15) * NFFT + bkoff;
    const unsigned short* bl_row =
        wlo + (size_t)(mat * NPAD + nc0 + lane15) * NFFT + bkoff;

    v8f acc[4] = {};   // 4 M-subtiles x 16x16 f32

    for (int k0 = 0; k0 < NFFT; k0 += 32) {
      // B fragment: lane holds column (nc0+lane15), K = k0+16*half .. +15
      v16bf bh = load_frag2(bh_row + k0, bh_row + k0 + 8);
      v16bf bl = load_frag2(bl_row + k0, bl_row + k0 + 8);

      // A fragment K offsets: contiguous K = 16*half + s, same as B
      const int offA0 = k0 + akoff;           // K {0..7} / {16..23}
      const int offA1 = offA0 + 8;            // K {8..15} / {24..31}
#pragma unroll
      for (int ms = 0; ms < 4; ++ms) {
        const unsigned short* a = arow[ms];
        v16bf ah = load_frag2(a + offA0, a + offA1);
        v16bf al = load_frag2(a + LDSROW + offA0, a + LDSROW + offA1);
        // split-float f32 emulation: hi*hi + hi*lo + lo*hi, f32 accumulate
        acc[ms] = __builtin_amdgcn_wmma_f32_16x16x32_bf16(
            false, ah, false, bh, (short)0, acc[ms], false, false);
        acc[ms] = __builtin_amdgcn_wmma_f32_16x16x32_bf16(
            false, ah, false, bl, (short)0, acc[ms], false, false);
        acc[ms] = __builtin_amdgcn_wmma_f32_16x16x32_bf16(
            false, al, false, bh, (short)0, acc[ms], false, false);
      }
    }

    // ---- store C/D: VGPR r -> row = r + 8*half, col = lane15 ----
    const int col = nc0 + lane15;
    if (col < NRFFT) {
      float* obase = out + (size_t)mat * ((size_t)MTOT * NRFFT);
#pragma unroll
      for (int ms = 0; ms < 4; ++ms) {
        const long row0 = m0 + ms * 16 + 8 * half;
#pragma unroll
        for (int r = 0; r < 8; ++r) {
          obase[(row0 + r) * (long)NRFFT + col] = acc[ms][r];
        }
      }
    }
  }
}

extern "C" void kernel_launch(void* const* d_in, const int* in_sizes, int n_in,
                              void* d_out, int out_size, void* d_ws, size_t ws_size,
                              hipStream_t stream) {
  (void)in_sizes; (void)n_in; (void)out_size; (void)ws_size;
  const float* x  = (const float*)d_in[0];
  const float* mr = (const float*)d_in[1];
  const float* mi = (const float*)d_in[2];
  float* out = (float*)d_out;

  unsigned short* whi = (unsigned short*)d_ws;                 // [2][272][512] bf16 hi
  unsigned short* wlo = whi + (size_t)2 * NPAD * NFFT;         // [2][272][512] bf16 lo

  const int wElems = 2 * NPAD * NFFT;                          // 278528
  split_weights<<<wElems / 256, 256, 0, stream>>>(mr, mi, whi, wlo);

  const size_t ldsBytes = (size_t)MTILE * RSTRIDE * sizeof(unsigned short); // 133120
  rfft_gemm<<<MTOT / MTILE, 256, ldsBytes, stream>>>(x, whi, wlo, out);
}
